// Baseline_64201171141020
// MI455X (gfx1250) — matrix operand order repaired and checked by reading, not compiled, not run on hardware
//
#include <hip/hip_runtime.h>

typedef __attribute__((ext_vector_type(16))) __bf16 v16bf;
typedef __attribute__((ext_vector_type(8)))  float  v8f;

#define HDIM 256
#define TDIM 512
#define MT   16     // batch rows per workgroup
#define SA   264    // padded LDS row stride (in elements) to dodge bank conflicts

// fast tanh: tanh(u) = 1 - 2/(exp(2u)+1) = 1 - 2*rcp(exp2(u*2/ln2)+1)
__device__ __forceinline__ float fast_tanh(float u) {
    float e = __builtin_amdgcn_exp2f(u * 2.885390082f);   // exp(2u)
    return 1.f - 2.f * __builtin_amdgcn_rcpf(e + 1.f);
}

__global__ __launch_bounds__(256)
void rnn_fused_kernel(const float* __restrict__ x,        // [B, T]
                      const float* __restrict__ W_embed,  // [1, H]
                      const float* __restrict__ b_embed,  // [H]
                      const float* __restrict__ W_update, // [H, H] (in, out)
                      const float* __restrict__ b_update, // [H]
                      const float* __restrict__ gamma,    // [H]
                      const float* __restrict__ beta,     // [H]
                      const float* __restrict__ W_out,    // [H, 10]
                      const float* __restrict__ b_out,    // [10]
                      float* __restrict__ out)            // [B, 10]
{
    __shared__ __bf16 sA[MT * SA];      // bf16 GEMM input tile (inp + h)
    __shared__ float  sR[MT * SA];      // raw tanh(GEMM) output (pre-LN)
    __shared__ float  sH[MT * SA];      // normalized hidden state
    __shared__ float  sX[MT * TDIM];    // this block's x rows (32 KB)
    __shared__ float  sGamma[HDIM];
    __shared__ float  sBeta[HDIM];

    const int tid   = threadIdx.x;          // 0..255
    const int lane  = tid & 31;
    const int wave  = tid >> 5;             // 0..7, owns N in [32w, 32w+32)
    const int lhalf = lane >> 4;            // 0/1 (lane group)
    const int l15   = lane & 15;
    const int row0  = blockIdx.x * MT;

    // ---- preload x rows (coalesced), gamma/beta, zero hidden state ----
    for (int idx = tid; idx < MT * TDIM; idx += 256) {
        int m = idx >> 9;               // /512
        int t = idx & (TDIM - 1);
        sX[idx] = x[(size_t)(row0 + m) * TDIM + t];
    }
    sGamma[tid] = gamma[tid];
    sBeta[tid]  = beta[tid];
    for (int idx = tid; idx < MT * SA; idx += 256) sH[idx] = 0.f;

    // per-thread embed params: this thread owns column n = tid for the A build
    const float we = W_embed[tid];
    const float be = b_embed[tid];

    // ---- W_update as register-resident bf16 B-fragments ----
    // B (32x16 KxN) layout: N = lane%16, element i -> K = 16*(lane/16) + i
    v16bf Bfrag[8][2];
    float cbias[2];
    for (int nt = 0; nt < 2; ++nt) {
        const int N = wave * 32 + nt * 16 + l15;
        cbias[nt] = b_update[N];
        for (int kc = 0; kc < 8; ++kc) {
            const int kbase = kc * 32 + lhalf * 16;
            v16bf f;
            #pragma unroll
            for (int i = 0; i < 16; ++i)
                f[i] = (__bf16)W_update[(size_t)(kbase + i) * HDIM + N];
            Bfrag[kc][nt] = f;
        }
    }

    const int lnRow = tid >> 4;   // layernorm row this thread helps with
    const int lnP   = tid & 15;   // its 16-column chunk

    __syncthreads();

    for (int t = 0; t < TDIM; ++t) {
        // ---- build A[m][n] = tanh(x[m,t]*We[n]+be[n]) + hN[m][n] (bf16) ----
        #pragma unroll 4
        for (int m = 0; m < MT; ++m) {
            float inp = fast_tanh(sX[m * TDIM + t] * we + be);
            sA[m * SA + tid] = (__bf16)(inp + sH[m * SA + tid]);
        }
        __syncthreads();

        // ---- GEMM: acc = b_update ; acc += A(16x256) @ Wupd(256x32 slice) ----
        v8f acc0, acc1;
        #pragma unroll
        for (int r = 0; r < 8; ++r) { acc0[r] = cbias[0]; acc1[r] = cbias[1]; }

        #pragma unroll
        for (int kc = 0; kc < 8; ++kc) {
            // A (16x32 MxK) fragment: M = lane%16,
            // elems 0..15 -> K = kc*32 + 16*lhalf + i
            const int kLo = kc * 32 + lhalf * 16;
            const __bf16* pLo = &sA[l15 * SA + kLo];       // 16B contiguous
            const __bf16* pHi = pLo + 8;                   // 16B contiguous
            v16bf a;
            #pragma unroll
            for (int i = 0; i < 8; ++i) { a[i] = pLo[i]; a[i + 8] = pHi[i]; }

            acc0 = __builtin_amdgcn_wmma_f32_16x16x32_bf16(
                       false, a, false, Bfrag[kc][0], (short)0, acc0, false, false);
            acc1 = __builtin_amdgcn_wmma_f32_16x16x32_bf16(
                       false, a, false, Bfrag[kc][1], (short)0, acc1, false, false);
        }

        // ---- tanh, scatter raw result to LDS (D layout: M = r + 8*lhalf) ----
        #pragma unroll
        for (int r = 0; r < 8; ++r) {
            const int m = r + lhalf * 8;
            sR[m * SA + wave * 32 +      l15] = fast_tanh(acc0[r]);
            sR[m * SA + wave * 32 + 16 + l15] = fast_tanh(acc1[r]);
        }
        __syncthreads();

        // ---- layernorm: 16 threads per row, shuffle-reduce within 16 lanes ----
        float s = 0.f, ss = 0.f;
        #pragma unroll
        for (int j = 0; j < 16; ++j) {
            float v = sR[lnRow * SA + lnP * 16 + j];
            s += v; ss += v * v;
        }
        #pragma unroll
        for (int msk = 1; msk < 16; msk <<= 1) {
            s  += __shfl_xor(s,  msk, 16);
            ss += __shfl_xor(ss, msk, 16);
        }
        const float mu   = s * (1.f / 256.f);
        const float var  = ss * (1.f / 256.f) - mu * mu;
        const float rstd = __builtin_amdgcn_rsqf(var + 1e-5f);
        #pragma unroll
        for (int j = 0; j < 16; ++j) {
            const int c = lnP * 16 + j;
            float v = sR[lnRow * SA + c];
            sH[lnRow * SA + c] = (v - mu) * rstd * sGamma[c] + sBeta[c];
        }
        __syncthreads();
    }

    // ---- output projection: h_final @ W_out + b_out  (16 rows x 10 cols) ----
    if (tid < MT * 10) {
        const int m = tid / 10, o = tid % 10;
        float acc = b_out[o];
        for (int k = 0; k < HDIM; ++k)
            acc += sH[m * SA + k] * W_out[k * 10 + o];
        out[(size_t)(row0 + m) * 10 + o] = acc;
    }
}

extern "C" void kernel_launch(void* const* d_in, const int* in_sizes, int n_in,
                              void* d_out, int out_size, void* d_ws, size_t ws_size,
                              hipStream_t stream) {
    const float* x        = (const float*)d_in[0];
    const float* W_embed  = (const float*)d_in[1];
    const float* b_embed  = (const float*)d_in[2];
    const float* W_update = (const float*)d_in[3];
    const float* b_update = (const float*)d_in[4];
    const float* gamma    = (const float*)d_in[5];
    const float* beta     = (const float*)d_in[6];
    const float* W_out    = (const float*)d_in[7];
    const float* b_out    = (const float*)d_in[8];
    float* out = (float*)d_out;

    const int B = in_sizes[0] / TDIM;          // 2048
    const int nblocks = B / MT;                // 128 persistent workgroups

    rnn_fused_kernel<<<nblocks, 256, 0, stream>>>(
        x, W_embed, b_embed, W_update, b_update, gamma, beta, W_out, b_out, out);
}
